// K_DAE_47820165874184
// MI455X (gfx1250) — matrix operand order repaired and checked by reading, not compiled, not run on hardware
//
#include <hip/hip_runtime.h>
#include <hip/hip_bf16.h>
#include <stdint.h>

typedef __attribute__((ext_vector_type(16))) __bf16 v16bf;
typedef __attribute__((ext_vector_type(8)))  __bf16 v8bf;
typedef __attribute__((ext_vector_type(8)))  float  v8f;

static constexpr int Bsz  = 16384;
static constexpr int Idim = 784;
static constexpr int Ipad = 800;    // 25 * 32, WMMA-K friendly
static constexpr int H1   = 512;
static constexpr int H2   = 256;
static constexpr int Ldim = 64;
static constexpr int Kc   = 10;     // clusters
static constexpr int BT   = 64;     // batch rows per workgroup
static constexpr int NW   = 8;      // waves per workgroup (256 threads, wave32)

// ---------------------------------------------------------------- conversion
__global__ void cvt_pad_x(const float* __restrict__ x, __bf16* __restrict__ xb) {
  long i = blockIdx.x * (long)blockDim.x + threadIdx.x;
  if (i >= (long)Bsz * Ipad) return;
  int  c = (int)(i % Ipad);
  long b = i / Ipad;
  xb[i] = (c < Idim) ? (__bf16)x[b * Idim + c] : (__bf16)0.0f;
}

// in:  [Kc][Kd][N] f32 row-major     out: [Kc][N][Kdpad] bf16 (K contiguous,
// zero-padded to Kdpad) -> B fragments become two contiguous 16B loads per lane.
__global__ void cvt_transpose(const float* __restrict__ in, __bf16* __restrict__ out,
                              int Kd, int Kdpad, int N, long total) {
  long i = blockIdx.x * (long)blockDim.x + threadIdx.x;
  if (i >= total) return;
  int  kd = (int)(i % Kdpad);
  long t  = i / Kdpad;
  int  n  = (int)(t % N);
  int  k  = (int)(t / N);
  out[i] = (kd < Kd) ? (__bf16)in[((long)k * Kd + kd) * N + n] : (__bf16)0.0f;
}

// ---------------------------------------------------------------- fragments
// A: 16x32 bf16, lane m = l%16, K base (l/16)*8; regs 0-3 = K[base..base+7],
// regs 4-7 = K[base+16..base+23]  -> two contiguous 16B loads per lane.
__device__ __forceinline__ v16bf load_a_frag(const __bf16* base, int lda,
                                             int m0, int k0, int lane) {
  int m  = m0 + (lane & 15);
  int kb = k0 + ((lane >> 4) << 3);
  const __bf16* p = base + (long)m * lda + kb;
  v8bf lo = *(const v8bf*)(p);
  v8bf hi = *(const v8bf*)(p + 16);
  v16bf r;
#pragma unroll
  for (int j = 0; j < 8; ++j) { r[j] = lo[j]; r[j + 8] = hi[j]; }
  return r;
}

// B: 32x16 bf16 from TRANSPOSED weights Wt[N][ldk]: lane n = l%16,
// K base (l/16)*8; regs 0-3 = K[base..base+7], regs 4-7 = K[base+16..base+23] (2x b128).
__device__ __forceinline__ v16bf load_b_frag_t(const __bf16* Wt, int ldk,
                                               int k0, int n0, int lane) {
  int n  = n0 + (lane & 15);
  int kk = k0 + ((lane >> 4) << 3);
  const __bf16* p = Wt + (long)n * ldk + kk;
  __builtin_prefetch((const void*)(p + 32), 0, 0);   // next K chunk (contiguous)
  v8bf lo = *(const v8bf*)(p);
  v8bf hi = *(const v8bf*)(p + 16);
  v16bf r;
#pragma unroll
  for (int j = 0; j < 8; ++j) { r[j] = lo[j]; r[j + 8] = hi[j]; }
  return r;
}

__device__ __forceinline__ void epilogue_store(const v8f* acc, int n0,
                                               const float* bias,
                                               __bf16* out, int ldo,
                                               bool relu, int lane) {
  const int   n    = n0 + (lane & 15);
  const float bv   = bias[n];
  const int   moff = (lane >> 4) << 3;   // C/D: lanes>=16 hold M = r+8
#pragma unroll
  for (int r = 0; r < 4; ++r) {
#pragma unroll
    for (int rr = 0; rr < 8; ++rr) {
      float v = acc[r][rr] + bv;
      if (relu) v = v > 0.0f ? v : 0.0f;
      out[(long)((r << 4) + moff + rr) * ldo + n] = (__bf16)v;
    }
  }
}

// One fused layer: out[BT x N] = act(A[BT x Kdim] * Wt^T + bias)
// Each wave owns 2 column tiles per pass; A fragments feed 2 WMMAs each.
__device__ __forceinline__ void gemm_stage(const __bf16* A, int lda,
                                           const __bf16* Wt, int ldk,
                                           int N, const float* bias,
                                           __bf16* out, int ldo,
                                           int Kdim, bool relu) {
  const int lane = threadIdx.x & 31;
  const int wave = threadIdx.x >> 5;
  for (int cp = wave; (cp << 5) < N; cp += NW) {
    const int  n0  = cp << 5;
    const bool two = (n0 + 16) < N;
    v8f acc0[4] = {};
    v8f acc1[4] = {};
    for (int kc = 0; kc < Kdim; kc += 32) {
      v16bf b0 = load_b_frag_t(Wt, ldk, kc, n0, lane);
      v16bf b1 = {};
      if (two) b1 = load_b_frag_t(Wt, ldk, kc, n0 + 16, lane);
#pragma unroll
      for (int r = 0; r < 4; ++r) {
        v16bf af = load_a_frag(A, lda, r << 4, kc, lane);
        acc0[r] = __builtin_amdgcn_wmma_f32_16x16x32_bf16(
            false, af, false, b0, (short)0, acc0[r], false, false);
        if (two)
          acc1[r] = __builtin_amdgcn_wmma_f32_16x16x32_bf16(
              false, af, false, b1, (short)0, acc1[r], false, false);
      }
    }
    epilogue_store(acc0, n0, bias, out, ldo, relu, lane);
    if (two) epilogue_store(acc1, n0 + 16, bias, out, ldo, relu, lane);
  }
}

__device__ __forceinline__ void final_epilogue(const v8f* acc, int n0,
                                               const float* bias,
                                               const float* __restrict__ xf,
                                               __bf16* __restrict__ recon,
                                               float* l2acc, long row0, int k,
                                               int lane) {
  const int   n    = n0 + (lane & 15);
  const float bv   = bias[n];
  const int   moff = (lane >> 4) << 3;
#pragma unroll
  for (int r = 0; r < 4; ++r) {
#pragma unroll
    for (int rr = 0; rr < 8; ++rr) {
      int   ml = (r << 4) + moff + rr;
      long  b  = row0 + ml;
      float v  = acc[r][rr] + bv;
      recon[((long)k * Bsz + b) * Idim + n] = (__bf16)v;
      float d = v - xf[b * (long)Idim + n];
      atomicAdd(&l2acc[ml], d * d);           // ds_add_f32
    }
  }
}

// Final layer: recon to global (bf16) + squared-L2 accumulation per row.
__device__ __forceinline__ void gemm_final(const __bf16* A, int lda,
                                           const __bf16* Wt, const float* bias,
                                           const float* __restrict__ xf,
                                           __bf16* __restrict__ recon,
                                           float* l2acc, long row0, int k) {
  const int lane = threadIdx.x & 31;
  const int wave = threadIdx.x >> 5;
  for (int cp = wave; (cp << 5) < Idim; cp += NW) {   // 25 passes (last single)
    const int  n0  = cp << 5;
    const bool two = (n0 + 16) < Idim;
    v8f acc0[4] = {};
    v8f acc1[4] = {};
    for (int kc = 0; kc < H1; kc += 32) {
      v16bf b0 = load_b_frag_t(Wt, H1, kc, n0, lane);
      v16bf b1 = {};
      if (two) b1 = load_b_frag_t(Wt, H1, kc, n0 + 16, lane);
#pragma unroll
      for (int r = 0; r < 4; ++r) {
        v16bf af = load_a_frag(A, lda, r << 4, kc, lane);
        acc0[r] = __builtin_amdgcn_wmma_f32_16x16x32_bf16(
            false, af, false, b0, (short)0, acc0[r], false, false);
        if (two)
          acc1[r] = __builtin_amdgcn_wmma_f32_16x16x32_bf16(
              false, af, false, b1, (short)0, acc1[r], false, false);
      }
    }
    final_epilogue(acc0, n0, bias, xf, recon, l2acc, row0, k, lane);
    if (two) final_epilogue(acc1, n0 + 16, bias, xf, recon, l2acc, row0, k, lane);
  }
}

// ---------------------------------------------------------------- fused fwd
__global__ __launch_bounds__(256) void fwd_kernel(
    const __bf16* __restrict__ xb,  const float* __restrict__ xf,
    const __bf16* __restrict__ we1, const float* __restrict__ be1,
    const __bf16* __restrict__ we2, const float* __restrict__ be2,
    const __bf16* __restrict__ we3, const float* __restrict__ be3,
    const __bf16* __restrict__ wd1, const float* __restrict__ bd1,
    const __bf16* __restrict__ wd2, const float* __restrict__ bd2,
    const __bf16* __restrict__ wd3, const float* __restrict__ bd3,
    __bf16* __restrict__ recon, float* __restrict__ l2) {
  __shared__ __align__(16) __bf16 sA[BT * H1];    // 64 KB
  __shared__ __align__(16) __bf16 sB[BT * H2];    // 32 KB
  __shared__ __align__(16) __bf16 sC[BT * Ldim];  //  8 KB
  __shared__ float l2acc[BT];

  const int  k    = blockIdx.y;
  const long row0 = (long)blockIdx.x * BT;

  for (int t = threadIdx.x; t < BT; t += blockDim.x) l2acc[t] = 0.0f;

  // L1: x[BT x 800] * We1^T  (Wt: [512][800]) -> sA
  gemm_stage(xb + row0 * Ipad, Ipad, we1 + (long)k * H1 * Ipad, Ipad,
             H1, be1 + (long)k * H1, sA, H1, Ipad, true);
  __syncthreads();
  // L2: sA * We2^T  (Wt: [256][512]) -> sB
  gemm_stage(sA, H1, we2 + (long)k * H2 * H1, H1,
             H2, be2 + (long)k * H2, sB, H2, H1, true);
  __syncthreads();
  // L3: sB * We3^T  (Wt: [64][256]) -> sC
  gemm_stage(sB, H2, we3 + (long)k * Ldim * H2, H2,
             Ldim, be3 + (long)k * Ldim, sC, Ldim, H2, true);
  __syncthreads();
  // L4: sC * Wd1^T  (Wt: [256][64]) -> sB
  gemm_stage(sC, Ldim, wd1 + (long)k * H2 * Ldim, Ldim,
             H2, bd1 + (long)k * H2, sB, H2, Ldim, true);
  __syncthreads();
  // L5: sB * Wd2^T  (Wt: [512][256]) -> sA
  gemm_stage(sB, H2, wd2 + (long)k * H1 * H2, H2,
             H1, bd2 + (long)k * H1, sA, H1, H2, true);
  __syncthreads();
  // L6: sA * Wd3^T  (Wt: [784][512]) -> recon (global) + L2 distance
  gemm_final(sA, H1, wd3 + (long)k * Idim * H1, bd3 + (long)k * Idim,
             xf, recon, l2acc, row0, k);
  __syncthreads();

  for (int t = threadIdx.x; t < BT; t += blockDim.x)
    l2[(row0 + t) * Kc + k] = l2acc[t];
}

// ---------------------------------------------------------------- argmin+gather
__global__ __launch_bounds__(128) void select_kernel(
    const float* __restrict__ l2, const __bf16* __restrict__ recon,
    float* __restrict__ out, int* __restrict__ outIdx) {
  const long b = blockIdx.x;
  int   kmin = 0;
  float best = l2[b * Kc];
#pragma unroll
  for (int k = 1; k < Kc; ++k) {
    float v = l2[b * Kc + k];
    if (v < best) { best = v; kmin = k; }   // first-min like jnp.argmin
  }
  const __bf16* src = recon + ((long)kmin * Bsz + b) * Idim;
  for (int c = threadIdx.x; c < Idim; c += blockDim.x)
    out[b * Idim + c] = (float)src[c];
  if (threadIdx.x == 0) outIdx[b] = kmin;
}

// ---------------------------------------------------------------- launch
extern "C" void kernel_launch(void* const* d_in, const int* in_sizes, int n_in,
                              void* d_out, int out_size, void* d_ws, size_t ws_size,
                              hipStream_t stream) {
  const float* x   = (const float*)d_in[0];
  const float* We1 = (const float*)d_in[1];
  const float* be1 = (const float*)d_in[2];
  const float* We2 = (const float*)d_in[3];
  const float* be2 = (const float*)d_in[4];
  const float* We3 = (const float*)d_in[5];
  const float* be3 = (const float*)d_in[6];
  const float* Wd1 = (const float*)d_in[7];
  const float* bd1 = (const float*)d_in[8];
  const float* Wd2 = (const float*)d_in[9];
  const float* bd2 = (const float*)d_in[10];
  const float* Wd3 = (const float*)d_in[11];
  const float* bd3 = (const float*)d_in[12];

  char*  ws  = (char*)d_ws;
  size_t off = 0;
  auto alloc = [&](size_t bytes) -> void* {
    void* p = ws + off;
    off += (bytes + 255) & ~(size_t)255;
    return p;
  };
  __bf16* xb    = (__bf16*)alloc((size_t)Bsz * Ipad * 2);
  __bf16* we1t  = (__bf16*)alloc((size_t)Kc * H1 * Ipad * 2);
  __bf16* we2t  = (__bf16*)alloc((size_t)Kc * H2 * H1 * 2);
  __bf16* we3t  = (__bf16*)alloc((size_t)Kc * Ldim * H2 * 2);
  __bf16* wd1t  = (__bf16*)alloc((size_t)Kc * H2 * Ldim * 2);
  __bf16* wd2t  = (__bf16*)alloc((size_t)Kc * H1 * H2 * 2);
  __bf16* wd3t  = (__bf16*)alloc((size_t)Kc * Idim * H1 * 2);
  __bf16* recon = (__bf16*)alloc((size_t)Kc * Bsz * Idim * 2);
  float*  l2    = (float*)alloc((size_t)Bsz * Kc * 4);
  (void)ws_size; (void)in_sizes; (void)n_in; (void)out_size;

  auto blocks = [](long n) { return (int)((n + 255) / 256); };

  cvt_pad_x<<<blocks((long)Bsz * Ipad), 256, 0, stream>>>(x, xb);
  // transposed (+K-padded) bf16 weights: Wt[k][N][Kdpad]
  cvt_transpose<<<blocks((long)Kc * H1 * Ipad), 256, 0, stream>>>(
      We1, we1t, Idim, Ipad, H1, (long)Kc * H1 * Ipad);
  cvt_transpose<<<blocks((long)Kc * H2 * H1), 256, 0, stream>>>(
      We2, we2t, H1, H1, H2, (long)Kc * H2 * H1);
  cvt_transpose<<<blocks((long)Kc * Ldim * H2), 256, 0, stream>>>(
      We3, we3t, H2, H2, Ldim, (long)Kc * Ldim * H2);
  cvt_transpose<<<blocks((long)Kc * H2 * Ldim), 256, 0, stream>>>(
      Wd1, wd1t, Ldim, Ldim, H2, (long)Kc * H2 * Ldim);
  cvt_transpose<<<blocks((long)Kc * H1 * H2), 256, 0, stream>>>(
      Wd2, wd2t, H2, H2, H1, (long)Kc * H1 * H2);
  cvt_transpose<<<blocks((long)Kc * Idim * H1), 256, 0, stream>>>(
      Wd3, wd3t, H1, H1, Idim, (long)Kc * Idim * H1);

  dim3 grid(Bsz / BT, Kc);
  fwd_kernel<<<grid, 256, 0, stream>>>(xb, x,
                                       we1t, be1, we2t, be2, we3t, be3,
                                       wd1t, bd1, wd2t, bd2, wd3t, bd3,
                                       recon, l2);

  float* selOut = (float*)d_out;
  int*   idxOut = (int*)(selOut + (size_t)Bsz * Idim);
  select_kernel<<<Bsz, 128, 0, stream>>>(l2, recon, selOut, idxOut);
}
